// STU_17205638987790
// MI455X (gfx1250) — matrix operand order repaired and checked by reading, not compiled, not run on hardware
//
#include <hip/hip_runtime.h>
#include <hip/hip_bf16.h>

// ---------------------------------------------------------------------------
// STU forward for MI455X (gfx1250, wave32, WMMA).
//  - conv + m_phi + AR: fused WMMA kernel, split bf16 hi/lo, f32 accumulate,
//    double-buffered async global->LDS staging (ASYNCcnt, in-order completion).
//  - y-recurrence: chunked affine scan (f16 M in LDS + v_dot2_f32_f16,
//    T^64 combine in f32).
// ---------------------------------------------------------------------------

typedef unsigned short u16_t;
typedef __attribute__((ext_vector_type(16))) __bf16         v16bf;
typedef __attribute__((ext_vector_type(16))) unsigned short v16u;
typedef __attribute__((ext_vector_type(8)))  float          v8f;
typedef __attribute__((ext_vector_type(2)))  _Float16       v2h_t;
typedef __attribute__((address_space(3)))    char           lds_char;

#define L_LEN 2048
#define D_DIM 256
#define K_EIG 24
#define KK_TOT 27          // 24 spectral filters + 3 AR taps share phase 2
#define B_SZ 4
#define WPAD 64
#define WROW (L_LEN + WPAD)   // 2112, zero-padded filter row
#define BM 64                 // t-rows per workgroup
#define NTB (L_LEN / BM)      // 32
#define S_STRIDE 40           // LDS stage row stride in halves (80B, 16B-aligned)
#define C_STRIDE 264          // LDS conv-tile row stride (bank stagger)
#define LC 64                 // scan chunk length (= 2^6 -> 6 squarings)
#define NCHUNK (L_LEN / LC)   // 32
#define MST 520               // scan LDS M row stride in halves (bank stagger)

#if __has_builtin(__builtin_amdgcn_fdot2)
#define HAVE_FDOT2 1
#else
#define HAVE_FDOT2 0
#endif

__device__ __forceinline__ u16_t f2bf(float f) {
  unsigned u = __float_as_uint(f);
  u += 0x7fffu + ((u >> 16) & 1u);          // round-to-nearest-even
  return (u16_t)(u >> 16);
}
__device__ __forceinline__ float bf2f(u16_t h) {
  return __uint_as_float(((unsigned)h) << 16);
}

__device__ __forceinline__ v8f wmma_bf16(v16u a, v16u b, v8f c) {
  return __builtin_amdgcn_wmma_f32_16x16x32_bf16(
      false, __builtin_bit_cast(v16bf, a),
      false, __builtin_bit_cast(v16bf, b),
      (short)0, c, false, false);
}

// Async global->LDS (CDNA5 VGLOBAL GLOBAL_LOAD_ASYNC_TO_LDS_B128, ASYNCcnt).
__device__ __forceinline__ void async_g2l_b128(lds_char* l, const void* g) {
  asm volatile("global_load_async_to_lds_b128 %0, %1, off"
               :: "v"(l), "v"(g) : "memory");
}
__device__ __forceinline__ void async_wait0() {
  asm volatile("s_wait_asynccnt 0" ::: "memory");
}
// One stage slab = 8 async instrs/wave; async loads complete in order, so
// waiting to level 8 retires the previous slab while the next stays in flight.
__device__ __forceinline__ void async_wait8() {
  asm volatile("s_wait_asynccnt 8" ::: "memory");
}

// stage u^T slab [d][s0..s0+31] (hi+lo): 4x16B per row per array
__device__ __forceinline__ void issue_stage_u(
    int tid, int b, int s0, const u16_t* uT_hi, const u16_t* uT_lo,
    lds_char* dh, lds_char* dl) {
  const char* gh = (const char*)uT_hi + ((((size_t)b << 19) + s0) << 1);
  const char* gl = (const char*)uT_lo + ((((size_t)b << 19) + s0) << 1);
#pragma unroll
  for (int i2 = 0; i2 < 4; ++i2) {
    int c2 = tid + (i2 << 8);
    int d = c2 >> 2, pb = (c2 & 3) << 4;
    async_g2l_b128(dh + d * (S_STRIDE * 2) + pb, gh + ((size_t)d << 12) + pb);
    async_g2l_b128(dl + d * (S_STRIDE * 2) + pb, gl + ((size_t)d << 12) + pb);
  }
}

// stage mT slab [o][dB*32 .. +31] (hi+lo)
__device__ __forceinline__ void issue_stage_m(
    int tid, int kk, int dB, const u16_t* mT_hi, const u16_t* mT_lo,
    lds_char* dh, lds_char* dl) {
  const char* gh = (const char*)mT_hi + ((size_t)kk << 17) + ((size_t)dB << 6);
  const char* gl = (const char*)mT_lo + ((size_t)kk << 17) + ((size_t)dB << 6);
#pragma unroll
  for (int i2 = 0; i2 < 4; ++i2) {
    int c2 = tid + (i2 << 8);
    int o = c2 >> 2, pb = (c2 & 3) << 4;
    async_g2l_b128(dh + o * (S_STRIDE * 2) + pb, gh + ((size_t)o << 9) + pb);
    async_g2l_b128(dl + o * (S_STRIDE * 2) + pb, gl + ((size_t)o << 9) + pb);
  }
}

// ---------------------------------------------------------------------------
// Prep kernels.
// ---------------------------------------------------------------------------

__global__ void prep_filters(const float* __restrict__ eig_vecs,
                             const float* __restrict__ eig_vals,
                             u16_t* __restrict__ w_hi, u16_t* __restrict__ w_lo) {
  int idx = blockIdx.x * blockDim.x + threadIdx.x;
  if (idx >= K_EIG * WROW) return;
  int k = idx / WROW, j = idx % WROW;
  float v = 0.f;
  if (j >= WPAD) {
    float ev = eig_vals[k];
    ev = ev > 0.f ? ev : 0.f;
    v = eig_vecs[(j - WPAD) * K_EIG + k] * sqrtf(sqrtf(ev));
  }
  u16_t h = f2bf(v);
  w_hi[idx] = h;
  w_lo[idx] = f2bf(v - bf2f(h));
}

__global__ void prep_u(const float* __restrict__ in,
                       u16_t* __restrict__ uN_hi, u16_t* __restrict__ uN_lo,
                       u16_t* __restrict__ uT_hi, u16_t* __restrict__ uT_lo) {
  int idx = blockIdx.x * blockDim.x + threadIdx.x;
  if (idx >= B_SZ * L_LEN * D_DIM) return;
  int d = idx & 255;
  int s = (idx >> 8) & 2047;
  int b = idx >> 19;
  float v = in[idx];
  u16_t h = f2bf(v), lo = f2bf(v - bf2f(h));
  uN_hi[idx] = h; uN_lo[idx] = lo;
  int ti = ((b << 8) + d) * L_LEN + s;
  uT_hi[ti] = h; uT_lo[ti] = lo;
}

__global__ void prep_mphi(const float* __restrict__ m_phi,
                          const float* __restrict__ m_u,
                          u16_t* __restrict__ mT_hi, u16_t* __restrict__ mT_lo) {
  int idx = blockIdx.x * blockDim.x + threadIdx.x;
  if (idx >= KK_TOT * D_DIM * D_DIM) return;
  int d = idx & 255, o = (idx >> 8) & 255, kk = idx >> 16;
  float v;
  if (kk < K_EIG) v = m_phi[((kk << 8) + d) * D_DIM + o];
  else            v = m_u[((o << 8) + d) * 3 + (kk - K_EIG)];
  u16_t h = f2bf(v);
  mT_hi[idx] = h;
  mT_lo[idx] = f2bf(v - bf2f(h));
}

__global__ void prep_mcatH(const float* __restrict__ m_y,
                           u16_t* __restrict__ McatH) {
  int idx = blockIdx.x * blockDim.x + threadIdx.x;
  if (idx >= 256 * 512) return;
  int dcat = idx & 511, o = idx >> 9;
  int j = dcat >> 8, dd = dcat & 255;
  _Float16 v = (_Float16)m_y[((o << 1) + j) * D_DIM + dd];
  McatH[idx] = __builtin_bit_cast(u16_t, v);
}

__global__ void build_T(const float* __restrict__ m_y, float* __restrict__ T) {
  int idx = blockIdx.x * blockDim.x + threadIdx.x;
  if (idx >= 512 * 512) return;
  int i = idx >> 9, j = idx & 511;
  float v;
  if (i < 256) v = (j < 256) ? m_y[((i << 1) + 0) * D_DIM + j]
                             : m_y[((i << 1) + 1) * D_DIM + (j - 256)];
  else         v = (j == i - 256) ? 1.f : 0.f;
  T[idx] = v;
}

__global__ __launch_bounds__(256) void sq_mm(const float* __restrict__ A,
                                             float* __restrict__ Cc) {
  __shared__ float As[32][33], Bs[32][33];
  const int bx = blockIdx.x, by = blockIdx.y;
  const int tid = threadIdx.x, tx = tid & 31, ty = tid >> 5;
  float acc[4] = {0.f, 0.f, 0.f, 0.f};
  for (int k0 = 0; k0 < 512; k0 += 32) {
    __syncthreads();
#pragma unroll
    for (int r = 0; r < 4; ++r) {
      int row = ty * 4 + r;
      As[row][tx] = A[((by << 5) + row) * 512 + k0 + tx];
      Bs[row][tx] = A[((k0 + row) << 9) + (bx << 5) + tx];
    }
    __syncthreads();
#pragma unroll
    for (int kk2 = 0; kk2 < 32; ++kk2) {
      float bv = Bs[kk2][tx];
#pragma unroll
      for (int r = 0; r < 4; ++r)
        acc[r] = fmaf(As[(ty << 2) + r][kk2], bv, acc[r]);
    }
  }
#pragma unroll
  for (int r = 0; r < 4; ++r)
    Cc[((by << 5) + (ty << 2) + r) * 512 + (bx << 5) + tx] = acc[r];
}

__global__ void transpose512(const float* __restrict__ A, float* __restrict__ At) {
  int idx = blockIdx.x * blockDim.x + threadIdx.x;
  if (idx >= 512 * 512) return;
  int i = idx >> 9, j = idx & 511;
  At[(j << 9) + i] = A[idx];
}

// ---------------------------------------------------------------------------
// Fused conv (+AR) + m_phi kernel.  grid=(tB=32, kGroup=3, b=4), block=256.
// Split precision: hi*hi + hi*lo + lo*hi on every WMMA.
// Filter slice stored reversed at BOTH half-word parities so every lane's
// Toeplitz gather is even-aligned (no v_mov_b16 reassembly).
// Double-buffered async staging: issue next slab, s_wait_asynccnt 8, barrier.
// ---------------------------------------------------------------------------
__global__ __launch_bounds__(256) void stu_conv(
    const u16_t* __restrict__ w_hi, const u16_t* __restrict__ w_lo,
    const u16_t* __restrict__ uT_hi, const u16_t* __restrict__ uT_lo,
    const u16_t* __restrict__ uN_hi, const u16_t* __restrict__ uN_lo,
    const u16_t* __restrict__ mT_hi, const u16_t* __restrict__ mT_lo,
    float* __restrict__ delta) {
  extern __shared__ char smem[];
  u16_t* stg_hi0 = (u16_t*)smem;                    // stage buf0 hi
  u16_t* stg_lo0 = stg_hi0 + 256 * S_STRIDE;
  u16_t* stg_hi1 = stg_lo0 + 256 * S_STRIDE;        // stage buf1 hi
  u16_t* stg_lo1 = stg_hi1 + 256 * S_STRIDE;
  u16_t* cb_hi   = stg_lo1 + 256 * S_STRIDE;        // [64][C_STRIDE]
  u16_t* cb_lo   = cb_hi + 64 * C_STRIDE;
  u16_t* wl_hi   = cb_lo + 64 * C_STRIDE;           // [128] reversed slice
  u16_t* wl_lo   = wl_hi + 128;
  u16_t* wlS_hi  = wl_lo + 128;                     // [128] shifted copy
  u16_t* wlS_lo  = wlS_hi + 128;
  lds_char* stg_hi0_l = (lds_char*)stg_hi0;
  lds_char* stg_lo0_l = (lds_char*)stg_lo0;
  lds_char* stg_hi1_l = (lds_char*)stg_hi1;
  lds_char* stg_lo1_l = (lds_char*)stg_lo1;

  const int tB = blockIdx.x, g = blockIdx.y, b = blockIdx.z;
  const int tBase = tB * BM;
  const int tid = threadIdx.x;
  const int lane = tid & 31, wv = tid >> 5;
  const int wt = wv >> 2, wo = wv & 3;
  const int hl = lane >> 4, m16 = lane & 15;

  v8f oacc[2][4];
#pragma unroll
  for (int i = 0; i < 2; ++i)
#pragma unroll
    for (int j = 0; j < 4; ++j) oacc[i][j] = (v8f){0,0,0,0,0,0,0,0};

  for (int kkl = 0; kkl < 9; ++kkl) {
    const int kk = g * 9 + kkl;

    if (kk < K_EIG) {
      // ---------------- phase 1: causal Toeplitz conv ----------------
      v8f cacc[2][4];
#pragma unroll
      for (int i = 0; i < 2; ++i)
#pragma unroll
        for (int j = 0; j < 4; ++j) cacc[i][j] = (v8f){0,0,0,0,0,0,0,0};

      const int nSB = 2 * tB + 2;                 // s < tBase + 64
      __syncthreads();                            // retire prior buf0 readers
      issue_stage_u(tid, b, 0, uT_hi, uT_lo, stg_hi0_l, stg_lo0_l);
      for (int sB = 0; sB < nSB; ++sB) {
        const int s0 = sB * 32;
        __syncthreads();                          // retire readers of buf[(sB+1)&1] / wl
        if (tid < 96) {                           // reversed slice + parity copy
          int wb = kk * WROW + (tBase - s0 + 33);
          u16_t vh = w_hi[wb + tid], vl = w_lo[wb + tid];
          int x = 95 - tid;
          wl_hi[x] = vh; wl_lo[x] = vl;
          if (x > 0) { wlS_hi[x - 1] = vh; wlS_lo[x - 1] = vl; }
        }
        if (sB + 1 < nSB) {
          issue_stage_u(tid, b, s0 + 32, uT_hi, uT_lo,
                        ((sB + 1) & 1) ? stg_hi1_l : stg_hi0_l,
                        ((sB + 1) & 1) ? stg_lo1_l : stg_lo0_l);
          async_wait8();                          // current slab landed
        } else {
          async_wait0();
        }
        __syncthreads();                          // publish
        const u16_t* sgh = (sB & 1) ? stg_hi1 : stg_hi0;
        const u16_t* sgl = (sB & 1) ? stg_lo1 : stg_lo0;

        // A fragments (Toeplitz): a[j] = wl[x0 + sj]; odd x0 -> shifted copy
        v16u ah[2], al[2];
#pragma unroll
        for (int mt = 0; mt < 2; ++mt) {
          int x0 = 64 - (32 * wt + 16 * mt + m16);
          int par = x0 & 1;
          const u16_t* ph = par ? wlS_hi : wl_hi;
          const u16_t* pl = par ? wlS_lo : wl_lo;
          int xb = x0 - par;
#pragma unroll
          for (int j = 0; j < 16; ++j) {
            int sj = j + 16 * hl;
            ah[mt][j] = ph[xb + sj];
            al[mt][j] = pl[xb + sj];
          }
        }
#pragma unroll
        for (int nt = 0; nt < 4; ++nt) {
          int d = 64 * wo + 16 * nt + m16;
          v16u bh, bl;
#pragma unroll
          for (int j = 0; j < 16; ++j) {          // b[j] = u[s0 + j + 16*hl, d]
            int sl = j + 16 * hl;
            bh[j] = sgh[d * S_STRIDE + sl];
            bl[j] = sgl[d * S_STRIDE + sl];
          }
#pragma unroll
          for (int mt = 0; mt < 2; ++mt) {
            cacc[mt][nt] = wmma_bf16(ah[mt], bh, cacc[mt][nt]);
            cacc[mt][nt] = wmma_bf16(ah[mt], bl, cacc[mt][nt]);
            cacc[mt][nt] = wmma_bf16(al[mt], bh, cacc[mt][nt]);
          }
        }
      }
      // spill conv tile to LDS as bf16 hi/lo for phase 2
      __syncthreads();
#pragma unroll
      for (int mt = 0; mt < 2; ++mt)
#pragma unroll
        for (int nt = 0; nt < 4; ++nt)
#pragma unroll
          for (int r = 0; r < 8; ++r) {
            int trow = 32 * wt + 16 * mt + r + 8 * hl;
            int d = 64 * wo + 16 * nt + m16;
            float v = cacc[mt][nt][r];
            u16_t h = f2bf(v);
            cb_hi[trow * C_STRIDE + d] = h;
            cb_lo[trow * C_STRIDE + d] = f2bf(v - bf2f(h));
          }
      __syncthreads();
    } else {
      // ---------------- phase 1 (AR): C = u shifted by iar ----------------
      const int iar = kk - K_EIG;
      __syncthreads();
      for (int e = tid; e < BM * D_DIM; e += 256) {
        int tl = e >> 8, d = e & 255;
        int tg = tBase + tl - iar;
        u16_t h = 0, lo = 0;
        if (tg >= 0) {
          int gi = ((b << 11) + tg) * D_DIM + d;
          h = uN_hi[gi]; lo = uN_lo[gi];
        }
        cb_hi[tl * C_STRIDE + d] = h;
        cb_lo[tl * C_STRIDE + d] = lo;
      }
      __syncthreads();
    }

    // ---------------- phase 2: Oacc += C @ mT[kk], pipelined over dB -------
    issue_stage_m(tid, kk, 0, mT_hi, mT_lo, stg_hi0_l, stg_lo0_l);
    for (int dB = 0; dB < 8; ++dB) {
      __syncthreads();                            // retire readers of buf[(dB+1)&1]
      if (dB + 1 < 8) {
        issue_stage_m(tid, kk, dB + 1, mT_hi, mT_lo,
                      ((dB + 1) & 1) ? stg_hi1_l : stg_hi0_l,
                      ((dB + 1) & 1) ? stg_lo1_l : stg_lo0_l);
        async_wait8();
      } else {
        async_wait0();
      }
      __syncthreads();                            // publish
      const u16_t* sgh = (dB & 1) ? stg_hi1 : stg_hi0;
      const u16_t* sgl = (dB & 1) ? stg_lo1 : stg_lo0;

      v16u ah[2], al[2];
#pragma unroll
      for (int mt = 0; mt < 2; ++mt) {
        int trow = 32 * wt + 16 * mt + m16;
#pragma unroll
        for (int j = 0; j < 16; ++j) {
          int dj = dB * 32 + j + 16 * hl;
          ah[mt][j] = cb_hi[trow * C_STRIDE + dj];
          al[mt][j] = cb_lo[trow * C_STRIDE + dj];
        }
      }
#pragma unroll
      for (int nt = 0; nt < 4; ++nt) {
        int o = 64 * wo + 16 * nt + m16;
        v16u bh, bl;
#pragma unroll
        for (int j = 0; j < 16; ++j) {
          int dl = j + 16 * hl;
          bh[j] = sgh[o * S_STRIDE + dl];
          bl[j] = sgl[o * S_STRIDE + dl];
        }
#pragma unroll
        for (int mt = 0; mt < 2; ++mt) {
          oacc[mt][nt] = wmma_bf16(ah[mt], bh, oacc[mt][nt]);
          oacc[mt][nt] = wmma_bf16(ah[mt], bl, oacc[mt][nt]);
          oacc[mt][nt] = wmma_bf16(al[mt], bh, oacc[mt][nt]);
        }
      }
    }
  } // kk loop

  // accumulate output tile into delta (3 k-groups -> 3 atomics per element)
#pragma unroll
  for (int mt = 0; mt < 2; ++mt)
#pragma unroll
    for (int nt = 0; nt < 4; ++nt)
#pragma unroll
      for (int r = 0; r < 8; ++r) {
        int t = tBase + 32 * wt + 16 * mt + r + 8 * hl;
        int o = 64 * wo + 16 * nt + m16;
        __hip_atomic_fetch_add(&delta[((b << 11) + t) * D_DIM + o],
                               oacc[mt][nt][r],
                               __ATOMIC_RELAXED, __HIP_MEMORY_SCOPE_AGENT);
      }
}

// ---------------------------------------------------------------------------
// Chunked scan, local pass.  grid=(NCHUNK, B), block=512.
// ---------------------------------------------------------------------------
__global__ __launch_bounds__(512) void stu_scan_chunk(
    const u16_t* __restrict__ McatH, const float* __restrict__ delta,
    const float* __restrict__ s_in, float* __restrict__ s_out,
    float* __restrict__ out, int pass) {
  extern __shared__ char smem[];
  u16_t* Mh  = (u16_t*)smem;                       // [256][MST] f16
  float* yb  = (float*)(smem + 256 * MST * 2);     // [2][256] f32 carry
  u16_t* yh  = (u16_t*)((char*)yb + 2 * 256 * 4);  // [2][256] f16 mirror
  float* pls = (float*)((char*)yh + 2 * 256 * 2);  // [256]

  const int c = blockIdx.x, b = blockIdx.y;
  const int tid = threadIdx.x, o = tid & 255, h = tid >> 8;

  for (int e = tid; e < 256 * 256; e += 512) {
    int oo = e >> 8, dw = e & 255;
    ((unsigned*)Mh)[oo * (MST / 2) + dw] = ((const unsigned*)McatH)[(oo << 8) + dw];
  }
  if (h == 0) {
    float y1 = 0.f, y2 = 0.f;
    if (pass == 1 && c > 0) {
      const float* sp = s_in + ((size_t)(b * NCHUNK + (c - 1))) * 512;
      y1 = sp[o]; y2 = sp[256 + o];
    }
    yb[256 + o] = y1; yb[o] = y2;                  // parity1=recent, parity0=older
    _Float16 h1 = (_Float16)y1, h2 = (_Float16)y2;
    yh[256 + o] = __builtin_bit_cast(u16_t, h1);
    yh[o]       = __builtin_bit_cast(u16_t, h2);
  }
  __syncthreads();

  const int t0 = c * LC;
  for (int tt = 0; tt < LC; ++tt) {
    const int srcp = (tt + 1 + h) & 1;
    float p = 0.f;
#if HAVE_FDOT2
    const v2h_t* mp = (const v2h_t*)&Mh[o * MST + (h << 8)];
    const v2h_t* yp = (const v2h_t*)&yh[srcp << 8];
#pragma unroll 8
    for (int q = 0; q < 128; ++q)
      p = __builtin_amdgcn_fdot2(mp[q], yp[q], p, false);
#else
    const u16_t* mp = &Mh[o * MST + (h << 8)];
    const float* yp = &yb[srcp << 8];
#pragma unroll 8
    for (int dd = 0; dd < 256; ++dd)
      p = fmaf((float)__builtin_bit_cast(_Float16, mp[dd]), yp[dd], p);
#endif
    if (h == 1) pls[o] = p;
    __syncthreads();
    if (h == 0) {
      int gi = (((b << 11) + t0 + tt) << 8) + o;
      float y = p + pls[o] + delta[gi];
      if (pass == 1) out[gi] = y;
      yb[((tt & 1) << 8) + o] = y;
      _Float16 yf = (_Float16)y;
      yh[((tt & 1) << 8) + o] = __builtin_bit_cast(u16_t, yf);
    }
    __syncthreads();
  }
  if (pass == 0 && h == 0) {                       // b_c = [y_last; y_last-1]
    float* sp = s_out + ((size_t)(b * NCHUNK + c)) * 512;
    sp[o]       = yb[256 + o];
    sp[256 + o] = yb[o];
  }
}

// s_c = A s_{c-1} + b_c, sequential over chunks.  grid=B, block=512.
__global__ __launch_bounds__(512) void stu_combine(
    const float* __restrict__ ATt, const float* __restrict__ bvec,
    float* __restrict__ svec) {
  __shared__ float s[512];
  const int b = blockIdx.x, i = threadIdx.x;
  s[i] = bvec[(size_t)(b * NCHUNK) * 512 + i];
  __syncthreads();
  svec[(size_t)(b * NCHUNK) * 512 + i] = s[i];
  for (int c = 1; c < NCHUNK; ++c) {
    float p = bvec[(size_t)(b * NCHUNK + c) * 512 + i];
#pragma unroll 4
    for (int j = 0; j < 512; ++j) p = fmaf(ATt[(j << 9) + i], s[j], p);
    __syncthreads();
    s[i] = p;
    svec[(size_t)(b * NCHUNK + c) * 512 + i] = p;
    __syncthreads();
  }
}

// ---------------------------------------------------------------------------
extern "C" void kernel_launch(void* const* d_in, const int* in_sizes, int n_in,
                              void* d_out, int out_size, void* d_ws, size_t ws_size,
                              hipStream_t stream) {
  const float* inputs   = (const float*)d_in[0];
  const float* eig_vals = (const float*)d_in[1];
  const float* eig_vecs = (const float*)d_in[2];
  const float* m_u      = (const float*)d_in[3];
  const float* m_phi    = (const float*)d_in[4];
  const float* m_y      = (const float*)d_in[5];
  float* out = (float*)d_out;

  char* p = (char*)d_ws;
  auto carve = [&](size_t bytes) -> char* {
    char* r = p;
    p += (bytes + 255) & ~(size_t)255;
    return r;
  };
  const size_t UE = (size_t)B_SZ * L_LEN * D_DIM;
  const size_t ME = (size_t)KK_TOT * D_DIM * D_DIM;
  u16_t* w_hi  = (u16_t*)carve((size_t)K_EIG * WROW * 2);
  u16_t* w_lo  = (u16_t*)carve((size_t)K_EIG * WROW * 2);
  u16_t* uT_hi = (u16_t*)carve(UE * 2);
  u16_t* uT_lo = (u16_t*)carve(UE * 2);
  u16_t* uN_hi = (u16_t*)carve(UE * 2);
  u16_t* uN_lo = (u16_t*)carve(UE * 2);
  u16_t* mT_hi = (u16_t*)carve(ME * 2);
  u16_t* mT_lo = (u16_t*)carve(ME * 2);
  float* delta = (float*)carve(UE * 4);
  u16_t* McatH = (u16_t*)carve((size_t)256 * 512 * 2);
  float* Pa    = (float*)carve((size_t)512 * 512 * 4);
  float* Pb    = (float*)carve((size_t)512 * 512 * 4);
  float* ATt   = (float*)carve((size_t)512 * 512 * 4);
  float* bvec  = (float*)carve((size_t)B_SZ * NCHUNK * 512 * 4);
  float* svec  = (float*)carve((size_t)B_SZ * NCHUNK * 512 * 4);

  hipMemsetAsync(delta, 0, UE * 4, stream);
  prep_filters<<<(K_EIG * WROW + 255) / 256, 256, 0, stream>>>(
      eig_vecs, eig_vals, w_hi, w_lo);
  prep_u<<<(int)((UE + 255) / 256), 256, 0, stream>>>(
      inputs, uN_hi, uN_lo, uT_hi, uT_lo);
  prep_mphi<<<(int)((ME + 255) / 256), 256, 0, stream>>>(
      m_phi, m_u, mT_hi, mT_lo);

  prep_mcatH<<<(256 * 512 + 255) / 256, 256, 0, stream>>>(m_y, McatH);
  build_T<<<(512 * 512 + 255) / 256, 256, 0, stream>>>(m_y, Pa);
  {
    float *pa = Pa, *pb = Pb;
    for (int it = 0; it < 6; ++it) {               // T^2 .. T^64
      sq_mm<<<dim3(16, 16), 256, 0, stream>>>(pa, pb);
      float* tswp = pa; pa = pb; pb = tswp;
    }
    transpose512<<<(512 * 512 + 255) / 256, 256, 0, stream>>>(pa, ATt);
  }

  size_t conv_lds =
      (size_t)(4 * 256 * S_STRIDE + 2 * 64 * C_STRIDE + 4 * 128) * sizeof(u16_t);
  stu_conv<<<dim3(NTB, 3, B_SZ), 256, conv_lds, stream>>>(
      w_hi, w_lo, uT_hi, uT_lo, uN_hi, uN_lo, mT_hi, mT_lo, delta);

  size_t scan_lds = (size_t)256 * MST * 2 + 2 * 256 * 4 + 2 * 256 * 2 + 256 * 4;
  stu_scan_chunk<<<dim3(NCHUNK, B_SZ), 512, scan_lds, stream>>>(
      McatH, delta, svec, bvec, out, 0);
  stu_combine<<<B_SZ, 512, 0, stream>>>(ATt, bvec, svec);
  stu_scan_chunk<<<dim3(NCHUNK, B_SZ), 512, scan_lds, stream>>>(
      McatH, delta, svec, bvec, out, 1);
}
